// LocalAttention_15530601742607
// MI455X (gfx1250) — compile-verified, matrix operand order checked by reading, not run on hardware
//
#include <hip/hip_runtime.h>

typedef __bf16 bf16_t;
typedef __attribute__((ext_vector_type(16))) __bf16 v16bf;
typedef __attribute__((ext_vector_type(8)))  float  v8f;
typedef __attribute__((ext_vector_type(4)))  unsigned int u32x4;
typedef __attribute__((ext_vector_type(8)))  int i32x8;
typedef __attribute__((ext_vector_type(4)))  int i32x4;

#define D_MODEL 1024
#define NHEAD   16
#define HEAD_DIM 64
#define WINDOW  128
#define BATCH   4
#define SEQ     4096
#define MTOT    (BATCH * SEQ)
#define NWIN    (SEQ / WINDOW)

// GEMM LDS: double-buffered A/B tiles, each [128][32] bf16 = 8 KB
#define LDS_TILE_BYTES 8192
#define LDS_GEMM_BYTES (4 * LDS_TILE_BYTES)   // A0 B0 A1 B1 = 32 KB

// Attention LDS carve (dynamic LDS assumed to start at LDS offset 0)
#define LDS_S   0                       // f32 [128][128]  64 KB
#define LDS_P   65536                   // bf16[128][128]  32 KB
#define LDS_VT  (65536 + 32768)         // bf16[64][128]   16 KB
#define LDS_KL  (LDS_VT + 16384)        // bf16[128][64]   16 KB
#define LDS_VL  (LDS_KL + 16384)        // bf16[128][64]   16 KB
#define LDS_ATTN_BYTES (LDS_VL + 16384) // 144 KB

#if defined(__has_builtin)
#if __has_builtin(__builtin_amdgcn_tensor_load_to_lds) && \
    __has_builtin(__builtin_amdgcn_s_wait_tensorcnt)
#define USE_TDM 1
#endif
#if __has_builtin(__builtin_amdgcn_global_load_async_to_lds_b128) && \
    __has_builtin(__builtin_amdgcn_s_wait_asynccnt)
#define USE_ASYNC_LDS 1
#endif
#endif

#if defined(USE_ASYNC_LDS)
// Builtin signature (from hipcc diagnostic): param0 is
// 'int __attribute__((vector_size(16))) __device__ *'  (= v4i addrspace(1)*),
// param1 is the LDS-side v4i addrspace(3)*.
typedef int gnu_v4i __attribute__((vector_size(16)));
typedef __attribute__((address_space(1))) gnu_v4i* g_v4i_ptr;
typedef __attribute__((address_space(3))) gnu_v4i* l_v4i_ptr;
// async 16-byte global -> LDS copy, tracked by ASYNCcnt (no VGPR data)
__device__ __forceinline__ void async_copy_b128(const void* g, void* l) {
    __builtin_amdgcn_global_load_async_to_lds_b128(
        (g_v4i_ptr)(uintptr_t)g, (l_v4i_ptr)(uintptr_t)l, 0, 0);
}
#endif

#if defined(USE_TDM)
// D# group 0: count=1 (valid), lds_addr, 57-bit global byte addr, type=2 (image)
__device__ __forceinline__ u32x4 tdm_group0(unsigned lds_byte_addr,
                                            unsigned long long gaddr) {
    u32x4 g;
    g.x = 1u;                                             // count=1
    g.y = lds_byte_addr;                                  // bits 63:32
    g.z = (unsigned)(gaddr & 0xFFFFFFFFu);                // bits 95:64
    g.w = (unsigned)((gaddr >> 32) & 0x01FFFFFFu) | 0x80000000u; // addr[56:32] | type=2
    return g;
}
// D# group 1: 2-byte elements, 2D tensor, 2D tile
__device__ __forceinline__ i32x8 tdm_group1_2d(unsigned tensor_d0, unsigned tensor_d1,
                                               unsigned tile_d0, unsigned tile_d1,
                                               unsigned long long stride0) {
    i32x8 g = {};
    g[0] = 0x10000;                                            // data_size=1 -> 2 bytes
    g[1] = (int)((tensor_d0 & 0xFFFFu) << 16);                 // tensor_dim0[15:0] @ 63:48
    g[2] = (int)((tensor_d0 >> 16) | ((tensor_d1 & 0xFFFFu) << 16)); // dim0 hi | dim1 lo
    g[3] = (int)((tensor_d1 >> 16) | (tile_d0 << 16));         // dim1 hi | tile_dim0
    g[4] = (int)(tile_d1 & 0xFFFFu);                           // tile_dim1 (tile_dim2=0)
    g[5] = (int)(stride0 & 0xFFFFFFFFu);                       // dim0_stride[31:0]
    g[6] = (int)((stride0 >> 32) & 0xFFFFu);                   // dim0_stride[47:32]
    g[7] = 0;
    return g;
}
__device__ __forceinline__ void tdm_load(u32x4 g0, i32x8 g1) {
    i32x4 z4 = {};
#if defined(__clang_major__) && __clang_major__ >= 23
    i32x8 z8 = {};
    __builtin_amdgcn_tensor_load_to_lds(g0, g1, z4, z4, z8, 0);
#else
    __builtin_amdgcn_tensor_load_to_lds(g0, g1, z4, z4, 0);
#endif
}
#endif

// ---------------------------------------------------------------------------
// Kernel 0: elementwise f32 -> bf16 (weight conversion)
// ---------------------------------------------------------------------------
__global__ void cvt_f32_to_bf16(const float* __restrict__ src,
                                bf16_t* __restrict__ dst, int n) {
    int i = blockIdx.x * blockDim.x + threadIdx.x;
    if (i < n) dst[i] = (bf16_t)src[i];
}

// ---------------------------------------------------------------------------
// Kernel 1: fused QKV projection.  out = x @ W^T + b   (W row-major [N][K])
// WG = 256 threads (8 wave32), tile M=128 N=128, K-step 32, double-buffered
// LDS with one barrier per step.  Next B tile is DMA'd global->LDS via
// async-load (ASYNCcnt); next A tile rides in 16 VGPRs and converts after
// the WMMA chain.  No fragment arrays -> no spills.
// ---------------------------------------------------------------------------
__global__ void __launch_bounds__(256)
qkv_proj_kernel(const float* __restrict__ xq,
                const float* __restrict__ xk,
                const float* __restrict__ xv,
                const bf16_t* __restrict__ Wqkv,   // [3][D][D]
                const float* __restrict__ bqkv,    // [3][D]
                bf16_t* __restrict__ qp,
                bf16_t* __restrict__ kp,
                bf16_t* __restrict__ vp) {
    extern __shared__ char smem_raw[];

    const int z = blockIdx.z;
    const float*  x    = (z == 0) ? xq : (z == 1) ? xk : xv;
    const bf16_t* w    = Wqkv + (size_t)z * D_MODEL * D_MODEL;
    const float*  bias = bqkv + z * D_MODEL;
    bf16_t*       outp = (z == 0) ? qp : (z == 1) ? kp : vp;

    const int tid  = threadIdx.x;
    const int lane = tid & 31;
    const int wave = tid >> 5;
    const int mBase = blockIdx.x * 128;
    const int nBase = blockIdx.y * 128;

    const int rowS  = tid >> 1;          // 0..127 staging row
    const int colS0 = (tid & 1) * 16;    // 0 or 16

    const int fragRow = lane & 15;
    const int fragK   = (lane >> 4) * 16;

    v8f acc[8] = {};

    const float*  gA = &x[(size_t)(mBase + rowS) * D_MODEL + colS0];
    const bf16_t* gB = &w[(size_t)(nBase + rowS) * D_MODEL + colS0];
    const int stageOff = rowS * 32 + colS0;      // bf16 elements within tile

    // stage tile 0 into buffer 0
    {
        bf16_t* dA = (bf16_t*)smem_raw + stageOff;
        bf16_t* dB = (bf16_t*)(smem_raw + LDS_TILE_BYTES) + stageOff;
#pragma unroll
        for (int j = 0; j < 16; ++j) dA[j] = (bf16_t)gA[j];
#if defined(USE_ASYNC_LDS)
        async_copy_b128(gB, dB);
        async_copy_b128(gB + 8, dB + 8);
        __builtin_amdgcn_s_wait_asynccnt(0);
#else
#pragma unroll
        for (int j = 0; j < 16; ++j) dB[j] = gB[j];
#endif
    }
    __syncthreads();

    for (int kk = 0; kk < D_MODEL; kk += 32) {
        const int buf = (kk >> 5) & 1;
        bf16_t* curA = (bf16_t*)(smem_raw + buf * 2 * LDS_TILE_BYTES);
        bf16_t* curB = (bf16_t*)(smem_raw + buf * 2 * LDS_TILE_BYTES + LDS_TILE_BYTES);
        bf16_t* nxtA = (bf16_t*)(smem_raw + (buf ^ 1) * 2 * LDS_TILE_BYTES);
        bf16_t* nxtB = (bf16_t*)(smem_raw + (buf ^ 1) * 2 * LDS_TILE_BYTES + LDS_TILE_BYTES);
        const bool hasNext = (kk + 32 < D_MODEL);

        // kick off next-tile movement before compute
        float aNext[16];
        if (hasNext) {
#if defined(USE_ASYNC_LDS)
            async_copy_b128(gB + kk + 32, nxtB + stageOff);       // DMA, no VGPRs
            async_copy_b128(gB + kk + 40, nxtB + stageOff + 8);
#endif
            __builtin_prefetch(gB + kk + 64, 0, 1);
            const float* sA = gA + kk + 32;
#pragma unroll
            for (int j = 0; j < 16; ++j) aNext[j] = sA[j];
        }

        // WMMA chain: one A fragment reused across 8 B tiles
        v16bf a = *(const v16bf*)&curA[(wave * 16 + fragRow) * 32 + fragK];
#pragma unroll
        for (int t = 0; t < 8; ++t) {
            v16bf b = *(const v16bf*)&curB[(t * 16 + fragRow) * 32 + fragK];
            acc[t] = __builtin_amdgcn_wmma_f32_16x16x32_bf16(
                false, a, false, b, (short)0, acc[t], false, false);
        }

        // commit next A tile (convert) and sync B fallback
        if (hasNext) {
            bf16_t* dA = nxtA + stageOff;
#pragma unroll
            for (int j = 0; j < 16; ++j) dA[j] = (bf16_t)aNext[j];
#if !defined(USE_ASYNC_LDS)
            const bf16_t* sB = gB + kk + 32;
            bf16_t* dB = nxtB + stageOff;
#pragma unroll
            for (int j = 0; j < 16; ++j) dB[j] = sB[j];
#endif
        }
#if defined(USE_ASYNC_LDS)
        __builtin_amdgcn_s_wait_asynccnt(0);
#endif
        __syncthreads();
    }

    const int halfSel = (lane >> 4) * 8;
    const int colSel  = lane & 15;
#pragma unroll
    for (int t = 0; t < 8; ++t) {
        const int colG = nBase + t * 16 + colSel;
        const float bv = bias[colG];
#pragma unroll
        for (int r = 0; r < 8; ++r) {
            const int rowG = mBase + wave * 16 + r + halfSel;
            outp[(size_t)rowG * D_MODEL + colG] = (bf16_t)(acc[t][r] + bv);
        }
    }
}

// ---------------------------------------------------------------------------
// Kernel 2: local windowed attention per (window, head, batch).
// K/V tiles DMA'd into LDS by the Tensor Data Mover (TENSORcnt path).
// ---------------------------------------------------------------------------
__global__ void __launch_bounds__(256)
local_attn_kernel(const bf16_t* __restrict__ qp,
                  const bf16_t* __restrict__ kp,
                  const bf16_t* __restrict__ vp,
                  bf16_t* __restrict__ outp) {
    extern __shared__ char smem_raw[];
    float*  S  = (float*)(smem_raw + LDS_S);    // [128][128] f32
    bf16_t* P  = (bf16_t*)(smem_raw + LDS_P);   // [128][128] bf16
    bf16_t* Vt = (bf16_t*)(smem_raw + LDS_VT);  // [64][128]  bf16
    bf16_t* Kl = (bf16_t*)(smem_raw + LDS_KL);  // [128][64]  bf16
    bf16_t* Vl = (bf16_t*)(smem_raw + LDS_VL);  // [128][64]  bf16

    const int win  = blockIdx.x;
    const int head = blockIdx.y;
    const int b    = blockIdx.z;
    const int tid  = threadIdx.x;
    const int lane = tid & 31;
    const int wave = tid >> 5;

    const size_t rowBase = (size_t)b * SEQ + (size_t)win * WINDOW;
    const int    colBase = head * HEAD_DIM;

#if defined(USE_TDM)
    // TDM: 2D tile [128 rows x 64 cols] of 2-byte elems, row stride 1024 elems.
    if (wave == 0) {
        unsigned long long kAddr =
            (unsigned long long)(uintptr_t)(kp + rowBase * D_MODEL + colBase);
        unsigned long long vAddr =
            (unsigned long long)(uintptr_t)(vp + rowBase * D_MODEL + colBase);
        i32x8 g1 = tdm_group1_2d(D_MODEL, MTOT, HEAD_DIM, WINDOW, D_MODEL);
        tdm_load(tdm_group0(LDS_KL, kAddr), g1);
        tdm_load(tdm_group0(LDS_VL, vAddr), g1);
        __builtin_amdgcn_s_wait_tensorcnt(0);
    }
    __syncthreads();
#else
    for (int i = tid; i < WINDOW * HEAD_DIM; i += 256) {
        int key = i >> 6;
        int hd  = i & 63;
        Kl[key * HEAD_DIM + hd] = kp[(rowBase + key) * D_MODEL + colBase + hd];
        Vl[key * HEAD_DIM + hd] = vp[(rowBase + key) * D_MODEL + colBase + hd];
    }
    __syncthreads();
#endif

    // transpose V: Vt[hd][key] (completes before the post-S barrier)
    for (int i = tid; i < WINDOW * HEAD_DIM; i += 256) {
        int key = i >> 6;
        int hd  = i & 63;
        Vt[hd * WINDOW + key] = Vl[key * HEAD_DIM + hd];
    }

    const int fragRow = lane & 15;
    const int fragK   = (lane >> 4) * 16;
    const int halfSel = (lane >> 4) * 8;
    const int colSel  = lane & 15;

    // S = (Q K^T) * 1/sqrt(64); each wave: 16 q-rows x 128 keys (8 N-tiles)
    v8f sacc[8] = {};
    for (int k0 = 0; k0 < HEAD_DIM; k0 += 32) {
        v16bf a = *(const v16bf*)&qp[(rowBase + wave * 16 + fragRow) * D_MODEL +
                                     colBase + k0 + fragK];
#pragma unroll
        for (int t = 0; t < 8; ++t) {
            v16bf bb = *(const v16bf*)&Kl[(t * 16 + fragRow) * HEAD_DIM + k0 + fragK];
            sacc[t] = __builtin_amdgcn_wmma_f32_16x16x32_bf16(
                false, a, false, bb, (short)0, sacc[t], false, false);
        }
    }
#pragma unroll
    for (int t = 0; t < 8; ++t) {
#pragma unroll
        for (int r = 0; r < 8; ++r)
            S[(wave * 16 + r + halfSel) * WINDOW + t * 16 + colSel] =
                sacc[t][r] * 0.125f;
    }
    __syncthreads();

    // row softmax -> bf16 P (128 rows, one thread each)
    if (tid < WINDOW) {
        float* row = &S[tid * WINDOW];
        float m = -1e30f;
        for (int j = 0; j < WINDOW; ++j) m = fmaxf(m, row[j]);
        float s = 0.f;
        for (int j = 0; j < WINDOW; ++j) {
            float e = __expf(row[j] - m);
            row[j] = e;
            s += e;
        }
        float inv = 1.f / s;
        bf16_t* prow = &P[tid * WINDOW];
        for (int j = 0; j < WINDOW; ++j) prow[j] = (bf16_t)(row[j] * inv);
    }
    __syncthreads();

    // O = P @ V; each wave: 16 q-rows x 64 hd, K over 128 keys
    v8f oacc[4] = {};
    for (int k0 = 0; k0 < WINDOW; k0 += 32) {
        v16bf a = *(const v16bf*)&P[(wave * 16 + fragRow) * WINDOW + k0 + fragK];
#pragma unroll
        for (int t = 0; t < 4; ++t) {
            v16bf bb = *(const v16bf*)&Vt[(t * 16 + fragRow) * WINDOW + k0 + fragK];
            oacc[t] = __builtin_amdgcn_wmma_f32_16x16x32_bf16(
                false, a, false, bb, (short)0, oacc[t], false, false);
        }
    }
#pragma unroll
    for (int t = 0; t < 4; ++t) {
#pragma unroll
        for (int r = 0; r < 8; ++r) {
            int rowG = wave * 16 + r + halfSel;
            outp[(rowBase + rowG) * D_MODEL + colBase + t * 16 + colSel] =
                (bf16_t)oacc[t][r];
        }
    }
}

// ---------------------------------------------------------------------------
// Kernel 3: output projection.  d_out = attn @ Wout^T + b  (f32 out)
// Both tiles are bf16 -> both staged by async global->LDS DMA.
// ---------------------------------------------------------------------------
__global__ void __launch_bounds__(256)
out_proj_kernel(const bf16_t* __restrict__ x,
                const bf16_t* __restrict__ w,
                const float* __restrict__ bias,
                float* __restrict__ out) {
    extern __shared__ char smem_raw[];

    const int tid  = threadIdx.x;
    const int lane = tid & 31;
    const int wave = tid >> 5;
    const int mBase = blockIdx.x * 128;
    const int nBase = blockIdx.y * 128;

    const int rowS  = tid >> 1;
    const int colS0 = (tid & 1) * 16;
    const int fragRow = lane & 15;
    const int fragK   = (lane >> 4) * 16;

    v8f acc[8] = {};

    const bf16_t* gA = &x[(size_t)(mBase + rowS) * D_MODEL + colS0];
    const bf16_t* gB = &w[(size_t)(nBase + rowS) * D_MODEL + colS0];
    const int stageOff = rowS * 32 + colS0;

    // stage tile 0 into buffer 0
    {
        bf16_t* dA = (bf16_t*)smem_raw + stageOff;
        bf16_t* dB = (bf16_t*)(smem_raw + LDS_TILE_BYTES) + stageOff;
#if defined(USE_ASYNC_LDS)
        async_copy_b128(gA, dA);
        async_copy_b128(gA + 8, dA + 8);
        async_copy_b128(gB, dB);
        async_copy_b128(gB + 8, dB + 8);
        __builtin_amdgcn_s_wait_asynccnt(0);
#else
#pragma unroll
        for (int j = 0; j < 16; ++j) dA[j] = gA[j];
#pragma unroll
        for (int j = 0; j < 16; ++j) dB[j] = gB[j];
#endif
    }
    __syncthreads();

    for (int kk = 0; kk < D_MODEL; kk += 32) {
        const int buf = (kk >> 5) & 1;
        bf16_t* curA = (bf16_t*)(smem_raw + buf * 2 * LDS_TILE_BYTES);
        bf16_t* curB = (bf16_t*)(smem_raw + buf * 2 * LDS_TILE_BYTES + LDS_TILE_BYTES);
        bf16_t* nxtA = (bf16_t*)(smem_raw + (buf ^ 1) * 2 * LDS_TILE_BYTES);
        bf16_t* nxtB = (bf16_t*)(smem_raw + (buf ^ 1) * 2 * LDS_TILE_BYTES + LDS_TILE_BYTES);
        const bool hasNext = (kk + 32 < D_MODEL);

        if (hasNext) {
            __builtin_prefetch(gB + kk + 64, 0, 1);
#if defined(USE_ASYNC_LDS)
            async_copy_b128(gA + kk + 32, nxtA + stageOff);
            async_copy_b128(gA + kk + 40, nxtA + stageOff + 8);
            async_copy_b128(gB + kk + 32, nxtB + stageOff);
            async_copy_b128(gB + kk + 40, nxtB + stageOff + 8);
#endif
        }

        v16bf a = *(const v16bf*)&curA[(wave * 16 + fragRow) * 32 + fragK];
#pragma unroll
        for (int t = 0; t < 8; ++t) {
            v16bf b = *(const v16bf*)&curB[(t * 16 + fragRow) * 32 + fragK];
            acc[t] = __builtin_amdgcn_wmma_f32_16x16x32_bf16(
                false, a, false, b, (short)0, acc[t], false, false);
        }

#if !defined(USE_ASYNC_LDS)
        if (hasNext) {
            const bf16_t* sA = gA + kk + 32;
            const bf16_t* sB = gB + kk + 32;
            bf16_t* dA = nxtA + stageOff;
            bf16_t* dB = nxtB + stageOff;
#pragma unroll
            for (int j = 0; j < 16; ++j) dA[j] = sA[j];
#pragma unroll
            for (int j = 0; j < 16; ++j) dB[j] = sB[j];
        }
#else
        __builtin_amdgcn_s_wait_asynccnt(0);
#endif
        __syncthreads();
    }

    const int halfSel = (lane >> 4) * 8;
    const int colSel  = lane & 15;
#pragma unroll
    for (int t = 0; t < 8; ++t) {
        const int colG = nBase + t * 16 + colSel;
        const float bv = bias[colG];
#pragma unroll
        for (int r = 0; r < 8; ++r) {
            const int rowG = mBase + wave * 16 + r + halfSel;
            out[(size_t)rowG * D_MODEL + colG] = acc[t][r] + bv;
        }
    }
}

// ---------------------------------------------------------------------------
extern "C" void kernel_launch(void* const* d_in, const int* in_sizes, int n_in,
                              void* d_out, int out_size, void* d_ws, size_t ws_size,
                              hipStream_t stream) {
    const float* q         = (const float*)d_in[0];
    const float* k         = (const float*)d_in[1];
    const float* v         = (const float*)d_in[2];
    const float* in_proj_w = (const float*)d_in[3];
    const float* in_proj_b = (const float*)d_in[4];
    const float* out_w     = (const float*)d_in[5];
    const float* out_b     = (const float*)d_in[6];
    float* out = (float*)d_out;

    char* ws = (char*)d_ws;
    const size_t szWqkv = (size_t)3 * D_MODEL * D_MODEL * sizeof(bf16_t); // 6 MB
    const size_t szWout = (size_t)D_MODEL * D_MODEL * sizeof(bf16_t);     // 2 MB
    const size_t szAct  = (size_t)MTOT * D_MODEL * sizeof(bf16_t);        // 32 MB

    bf16_t* Wqkv = (bf16_t*)ws;
    bf16_t* Wout = (bf16_t*)(ws + szWqkv);
    bf16_t* qp   = (bf16_t*)(ws + szWqkv + szWout);
    bf16_t* kp   = (bf16_t*)(ws + szWqkv + szWout + szAct);
    bf16_t* vp   = (bf16_t*)(ws + szWqkv + szWout + 2 * szAct);
    bf16_t* attn = (bf16_t*)(ws + szWqkv + szWout + 3 * szAct);

    // 0) weight conversion
    {
        int n1 = 3 * D_MODEL * D_MODEL;
        cvt_f32_to_bf16<<<(n1 + 255) / 256, 256, 0, stream>>>(in_proj_w, Wqkv, n1);
        int n2 = D_MODEL * D_MODEL;
        cvt_f32_to_bf16<<<(n2 + 255) / 256, 256, 0, stream>>>(out_w, Wout, n2);
    }

    // 1) QKV projections (z selects q/k/v)
    {
        dim3 grid(MTOT / 128, D_MODEL / 128, 3);
        qkv_proj_kernel<<<grid, 256, LDS_GEMM_BYTES, stream>>>(q, k, v, Wqkv,
                                                               in_proj_b, qp, kp, vp);
    }

    // 2) windowed attention (TDM staging of K/V tiles)
    {
        dim3 grid(NWIN, NHEAD, BATCH);
        local_attn_kernel<<<grid, 256, LDS_ATTN_BYTES, stream>>>(qp, kp, vp, attn);
    }

    // 3) output projection
    {
        dim3 grid(MTOT / 128, D_MODEL / 128);
        out_proj_kernel<<<grid, 256, LDS_GEMM_BYTES, stream>>>(attn, Wout, out_b, out);
    }
}
